// DynamicPerPixelConv2d_1529008358155
// MI455X (gfx1250) — matrix operand order repaired and checked by reading, not compiled, not run on hardware
//
#include <hip/hip_runtime.h>

typedef __attribute__((ext_vector_type(16))) _Float16 v16h;
typedef __attribute__((ext_vector_type(8)))  float    v8f;
typedef __attribute__((ext_vector_type(4)))  float    v4f;

#define BB      2
#define CIN     32
#define HH      64
#define WW      64
#define COUT    32
#define KK      9
#define PRED_W  9216
#define ROWS_O  288          // CIN*KK rows of Wp per output channel
#define NPIX    16           // pixel tile = WMMA N
#define PSTRIDE 292          // 288 + 4 pad -> conflict-free b128 stores
#define NWAVES  2

__global__ void wp_to_f16_kernel(const float* __restrict__ Wp,
                                 _Float16* __restrict__ Wp16, int n) {
  int i = blockIdx.x * blockDim.x + threadIdx.x;
  if (i < n) Wp16[i] = (_Float16)Wp[i];
}

__global__ __launch_bounds__(64)
void dynconv_fused_kernel(const float* __restrict__ x,
                          const float* __restrict__ Wp,      // f32, for bias row
                          const _Float16* __restrict__ Wp16, // preconverted
                          const float* __restrict__ bp,
                          float* __restrict__ out)
{
  __shared__ __align__(16) float predbuf[NWAVES][NPIX * PSTRIDE]; // wave-private
  __shared__ __align__(16) float patch[NPIX * PSTRIDE];
  __shared__ __align__(16) float xtile[CIN * NPIX];

  const int tid  = threadIdx.x;
  const int wave = tid >> 5;
  const int lane = tid & 31;
  const int n    = lane & 15;   // pixel within tile / WMMA N
  const int hi   = lane >> 4;   // half-wave selector

  const int wg  = blockIdx.x;          // 512 WGs: 256 tiles per batch image
  const int b   = wg >> 8;
  const int rem = wg & 255;
  const int h   = rem >> 2;
  const int w0  = (rem & 3) << 4;

  // ---- stage x tile (32 cin x 16 pixels) ----
  for (int i = tid; i < CIN * NPIX; i += 64) {
    int c = i >> 4, nn = i & 15;
    xtile[i] = x[((b * CIN + c) * HH + h) * WW + w0 + nn];
  }
  // ---- stage 3x3 patches: patch[n][c*9+k], zero padded ----
  for (int i = tid; i < NPIX * ROWS_O; i += 64) {
    int nn = i / ROWS_O;
    int r  = i - nn * ROWS_O;
    int c  = r / KK;
    int k  = r - c * KK;
    int kh = k / 3, kw = k - kh * 3;
    int h2 = h + kh - 1;
    int w2 = w0 + nn + kw - 1;
    float v = 0.f;
    if (h2 >= 0 && h2 < HH && w2 >= 0 && w2 < WW)
      v = x[((b * CIN + c) * HH + h2) * WW + w2];
    patch[nn * PSTRIDE + r] = v;
  }
  __syncthreads();

  // ---- B fragment (K=32 x N=16, f16): lanes 0-15 hold K 0..15, 16-31 hold K 16..31
  v16h bfrag;
#pragma unroll
  for (int e = 0; e < 16; ++e) {
    int kidx = hi * 16 + e;
    bfrag[e] = (_Float16)xtile[kidx * NPIX + n];
  }

  float* pw = predbuf[wave];

  for (int o = wave; o < COUT; o += NWAVES) {
    const int rowbase = o * ROWS_O;

    // ---- 18 WMMA m-tiles: pred rows [o*288, o*288+288) x 16 pixels ----
    for (int m = 0; m < 18; ++m) {
      // A: lane row = rowbase + m*16 + (lane%16); contiguous K-halves, same order as B
      const _Float16* arow = Wp16 + (rowbase + m * 16 + n) * CIN;
      union { v4f f[2]; v16h h; } af;
      const v4f* ap = (const v4f*)arow;       // 64B row, 16B chunks
      af.f[0] = ap[2 * hi];                   // K = 16hi .. 16hi+7
      af.f[1] = ap[2 * hi + 1];               // K = 16hi+8 .. 16hi+15

      // C: broadcast predictor bias bp[row] per D row (row = m*16 + 8hi + v)
      union { v4f f[2]; v8f v; } cf;
      const float* bprow = bp + rowbase + m * 16 + 8 * hi;
      cf.f[0] = *(const v4f*)(bprow);
      cf.f[1] = *(const v4f*)(bprow + 4);

      v8f d = __builtin_amdgcn_wmma_f32_16x16x32_f16(
          false, af.h, false, bfrag, (short)0, cf.v, false, false);

      // store pixel-major: pred[n][m*16 + 8hi + v], two b128, bank-conflict-free
      union { v8f v; v4f f[2]; } du; du.v = d;
      float* dst = pw + n * PSTRIDE + m * 16 + 8 * hi;
      *(v4f*)dst       = du.f[0];
      *(v4f*)(dst + 4) = du.f[1];
    }

    // ---- per-pixel normalize + contract: 2 lanes per pixel (split cin) ----
    const float* wpx = pw + n * PSTRIDE;
    const float* ptx = patch + n * PSTRIDE;

    float s[KK];
#pragma unroll
    for (int k = 0; k < KK; ++k) s[k] = 0.f;
    for (int c = 0; c < 16; ++c) {
      const float* wr = wpx + (hi * 16 + c) * KK;
#pragma unroll
      for (int k = 0; k < KK; ++k) { float w = wr[k]; s[k] += w * w; }
    }
#pragma unroll
    for (int k = 0; k < KK; ++k) s[k] += __shfl_xor(s[k], 16, 32);
    float rs[KK];
#pragma unroll
    for (int k = 0; k < KK; ++k) rs[k] = 1.0f / fmaxf(sqrtf(s[k]), 1e-12f);

    float acc = 0.f;
    for (int c = 0; c < 16; ++c) {
      int cc = hi * 16 + c;
      const float* wr = wpx + cc * KK;
      float wv[KK];
      float t = 0.f;
#pragma unroll
      for (int k = 0; k < KK; ++k) { wv[k] = wr[k] * rs[k]; t += wv[k] * wv[k]; }
      float rt = 1.0f / fmaxf(sqrtf(t), 1e-12f);
      const float* pr = ptx + cc * KK;
#pragma unroll
      for (int k = 0; k < KK; ++k) acc += wv[k] * rt * pr[k];
    }

    // per-pixel dynamic bias row (exact f32)
    float bia = 0.f;
    const float* wbias = Wp + (PRED_W + o) * CIN;
#pragma unroll
    for (int c = 0; c < 16; ++c) {
      int cc = hi * 16 + c;
      bia += wbias[cc] * xtile[cc * NPIX + n];
    }

    float tot = acc + bia;
    tot += __shfl_xor(tot, 16, 32);
    if (hi == 0)
      out[(b * COUT + o) * (HH * WW) + h * WW + w0 + n] = tot + bp[PRED_W + o];
  }
}

extern "C" void kernel_launch(void* const* d_in, const int* in_sizes, int n_in,
                              void* d_out, int out_size, void* d_ws, size_t ws_size,
                              hipStream_t stream) {
  const float* x  = (const float*)d_in[0];
  const float* Wp = (const float*)d_in[1];
  const float* bp = (const float*)d_in[2];
  float* out = (float*)d_out;

  _Float16* Wp16 = (_Float16*)d_ws;   // 9248*32*2 = 578 KB of scratch
  const int nw = (PRED_W + COUT) * CIN;

  wp_to_f16_kernel<<<(nw + 255) / 256, 256, 0, stream>>>(Wp, Wp16, nw);
  dynconv_fused_kernel<<<BB * (HH * WW) / NPIX, 64, 0, stream>>>(x, Wp, Wp16, bp, out);
}
